// GeographyAwareMultiHeadAttention_21672404975718
// MI455X (gfx1250) — compile-verified, matrix operand order checked by reading, not run on hardware
//
#include <hip/hip_runtime.h>

#define BT      128
#define NCELLS  512
#define DMODEL  256
#define NHEADS  8
#define HDIM    32

typedef __bf16 bf16;
typedef __attribute__((ext_vector_type(16))) __bf16 v16bf;
typedef __attribute__((ext_vector_type(8)))  __bf16 v8bf;
typedef __attribute__((ext_vector_type(8)))  float  v8f;
typedef __attribute__((ext_vector_type(4)))  float  v4f;

__device__ __forceinline__ bf16 f2bf(float f) {
  unsigned u = __builtin_bit_cast(unsigned, f);
  unsigned r = (u + 0x7FFFu + ((u >> 16) & 1u)) >> 16;
  unsigned short h = (unsigned short)r;
  return __builtin_bit_cast(bf16, h);
}

// A/B fragment for v_wmma_f32_16x16x32_bf16:
// lane L (row/col = L%16) holds K = {b..b+7} U {b+16..b+23}, b = 8*(L>=16).
// Caller passes the two contiguous 8-element (16 B) runs; concat via a single
// shufflevector so it coalesces into the fragment VGPRs with no v_mov chains.
__device__ __forceinline__ v16bf frag_from_runs(const bf16* p0, const bf16* p1) {
  v8bf lo = *(const v8bf*)p0;
  v8bf hi = *(const v8bf*)p1;
  return __builtin_shufflevector(lo, hi, 0, 1, 2, 3, 4, 5, 6, 7,
                                 8, 9, 10, 11, 12, 13, 14, 15);
}

__device__ __forceinline__ v8f zero8() {
  v8f z;
#pragma unroll
  for (int i = 0; i < 8; ++i) z[i] = 0.0f;
  return z;
}

// ---------------------------------------------------------------------------
// Kernel 0: one-shot f32 -> bf16 conversion of x (8 elements / thread).
// ---------------------------------------------------------------------------
__global__ void conv_x(const float* __restrict__ x, bf16* __restrict__ xb) {
  size_t i = ((size_t)blockIdx.x * blockDim.x + threadIdx.x) * 8;
  v4f a = *(const v4f*)(x + i);
  v4f b = *(const v4f*)(x + i + 4);
  v8bf o;
#pragma unroll
  for (int j = 0; j < 4; ++j) { o[j] = f2bf(a[j]); o[j + 4] = f2bf(b[j]); }
  *(v8bf*)(xb + i) = o;
}

// ---------------------------------------------------------------------------
// Kernel 1: convert + transpose weights to bf16.  WT[mat][n][k] = W[mat][k][n]
// ---------------------------------------------------------------------------
__global__ void prep_weights(const float* Wq, const float* Wk, const float* Wv,
                             const float* Wo, bf16* WT /* [4][256][256] */) {
  int idx = blockIdx.x * blockDim.x + threadIdx.x;   // 0 .. 4*65536-1
  int mat = idx >> 16;
  int e   = idx & 65535;
  int n   = e >> 8;
  int k   = e & 255;
  const float* W = (mat == 0) ? Wq : (mat == 1) ? Wk : (mat == 2) ? Wv : Wo;
  WT[(size_t)mat * DMODEL * DMODEL + n * DMODEL + k] = f2bf(W[k * DMODEL + n]);
}

// ---------------------------------------------------------------------------
// Kernel 2: fused QKV projection.  One 16x16 C-tile per wave, K=256 -> 8 WMMAs.
// A and B fragments are pure b128 loads (x pre-converted to bf16).
// q,k stored [bt,h,cell,hd]; v stored transposed [bt,h,hd,cell].
// ---------------------------------------------------------------------------
__global__ void qkv_proj(const bf16* __restrict__ xb, const bf16* __restrict__ WT,
                         const float* __restrict__ bq, const float* __restrict__ bk,
                         const float* __restrict__ bv,
                         bf16* __restrict__ qb, bf16* __restrict__ kb,
                         bf16* __restrict__ vtb) {
  int wave = threadIdx.x >> 5;
  int lane = threadIdx.x & 31;
  int tile = blockIdx.x * 4 + wave;      // 0 .. 3*65536-1
  int mat  = tile >> 16;                 // 0=q 1=k 2=v
  int rem  = tile & 65535;
  int rowtile = rem >> 4;
  int coltile = rem & 15;
  int row0 = rowtile * 16;
  int lrow = lane & 15;
  int hi   = lane >> 4;
  int kb8  = hi * 8;
  int col  = coltile * 16 + lrow;

  const bf16* Wt   = WT + (size_t)mat * DMODEL * DMODEL + (size_t)col * DMODEL;
  const bf16* xrow = xb + (size_t)(row0 + lrow) * DMODEL;

  v8f acc = zero8();
#pragma unroll
  for (int kc = 0; kc < 8; ++kc) {
    int k0 = kc * 32 + kb8;
    v16bf a = frag_from_runs(xrow + k0, xrow + k0 + 16);
    v16bf b = frag_from_runs(Wt + k0, Wt + k0 + 16);
    acc = __builtin_amdgcn_wmma_f32_16x16x32_bf16(false, a, false, b,
                                                  (short)0, acc, false, false);
  }
  const float* bias = (mat == 0) ? bq : (mat == 1) ? bk : bv;
  float bval = bias[col];
  int h = col >> 5;
  int d = col & 31;
  if (mat == 2) {
    v8bf pk;
#pragma unroll
    for (int r = 0; r < 8; ++r) pk[r] = f2bf(acc[r] + bval);
    int grow = row0 + 8 * hi;            // 8 consecutive rows from here
    int btI  = grow >> 9;
    int cell = grow & 511;
    bf16* dst = vtb + ((size_t)((btI * NHEADS + h) * HDIM + d)) * NCELLS + cell;
    *(v8bf*)dst = pk;
  } else {
    bf16* base = (mat == 0) ? qb : kb;
#pragma unroll
    for (int r = 0; r < 8; ++r) {
      int grow = row0 + r + 8 * hi;
      int btI  = grow >> 9;
      int cell = grow & 511;
      base[((size_t)((btI * NHEADS + h) * NCELLS + cell)) * HDIM + d] =
          f2bf(acc[r] + bval);
    }
  }
}

// ---------------------------------------------------------------------------
// Kernel 3: attention.  One workgroup (4 waves) per (bt, h, 16-row q-tile).
// hd=32 -> each 16x16 logit tile is exactly one bf16 WMMA.
// The spatial-constraint bias tile (16x512 f32, contiguous 32 KB) is staged
// into LDS with gfx1250 async-to-LDS copies overlapping the Q fragment load.
// ---------------------------------------------------------------------------
__global__ void attn_kernel(const bf16* __restrict__ qb, const bf16* __restrict__ kb,
                            const bf16* __restrict__ vtb, const float* __restrict__ sc,
                            bf16* __restrict__ att) {
  __shared__ float scL[16][NCELLS];                  // 32 KB bias/mask tile
  __shared__ float lg[16][NCELLS];                   // 32 KB logits
  __shared__ __align__(16) bf16 pr[16][NCELLS];      // 16 KB probs (bf16)
  __shared__ float part[4][16][16];                  //  4 KB partial attn*V

  int wave = threadIdx.x >> 5;
  int lane = threadIdx.x & 31;
  int lrow = lane & 15;
  int hi   = lane >> 4;
  int kb8  = hi * 8;

  int blk   = blockIdx.x;                 // (bt*H + h)*32 + ntile
  int ntile = blk & 31;
  int bh    = blk >> 5;
  int n0    = ntile * 16;

  // --- async bulk copy: sc[n0:n0+16, :] (contiguous 32 KB) -> LDS ---------
  {
    unsigned lbase = (unsigned)(size_t)&scL[0][0] + threadIdx.x * 256u;
    unsigned long long gbase =
        (unsigned long long)(sc + (size_t)n0 * NCELLS) + threadIdx.x * 256ull;
#pragma unroll
    for (int j = 0; j < 16; ++j) {
      asm volatile("global_load_async_to_lds_b128 %0, %1, off offset:%2"
                   :: "v"(lbase), "v"(gbase), "i"(j * 16) : "memory");
    }
  }

  const bf16* qh = qb  + (size_t)bh * NCELLS * HDIM;
  const bf16* kh = kb  + (size_t)bh * NCELLS * HDIM;
  const bf16* vh = vtb + (size_t)bh * HDIM * NCELLS;

  // A fragment: 16 q rows, full hd=32 (overlaps with the async copies)
  v16bf aq = frag_from_runs(qh + (size_t)(n0 + lrow) * HDIM + kb8,
                            qh + (size_t)(n0 + lrow) * HDIM + kb8 + 16);

  // all async copies done (per wave) + visible to whole workgroup
  asm volatile("s_wait_asynccnt 0x0" ::: "memory");
  __syncthreads();

  const float scale = 0.17677669529663687f;          // 1/sqrt(32)

  // --- QK^T: 32 column tiles, 8 per wave, 1 WMMA each ---
  for (int i = 0; i < 8; ++i) {
    int mt = wave + 4 * i;
    int m  = mt * 16 + lrow;
    v16bf bfr = frag_from_runs(kh + (size_t)m * HDIM + kb8,
                               kh + (size_t)m * HDIM + kb8 + 16);
    v8f c = zero8();
    c = __builtin_amdgcn_wmma_f32_16x16x32_bf16(false, aq, false, bfr,
                                                (short)0, c, false, false);
#pragma unroll
    for (int r = 0; r < 8; ++r) {
      int nloc = r + 8 * hi;
      float s = scL[nloc][m];
      lg[nloc][m] = (s > 0.0f) ? (c[r] * scale + s) : -1e9f;
    }
  }
  __syncthreads();

  // --- softmax: 8 lanes per row (shfl_xor within lane groups of 8) ---
  {
    int row = threadIdx.x >> 3;
    int seg = threadIdx.x & 7;
    float mx = -3.0e38f;
    for (int c = seg * 64; c < seg * 64 + 64; ++c) mx = fmaxf(mx, lg[row][c]);
    mx = fmaxf(mx, __shfl_xor(mx, 1, 8));
    mx = fmaxf(mx, __shfl_xor(mx, 2, 8));
    mx = fmaxf(mx, __shfl_xor(mx, 4, 8));
    float sum = 0.0f;
    for (int c = seg * 64; c < seg * 64 + 64; ++c) {
      float e = __expf(lg[row][c] - mx);
      lg[row][c] = e;
      sum += e;
    }
    sum += __shfl_xor(sum, 1, 8);
    sum += __shfl_xor(sum, 2, 8);
    sum += __shfl_xor(sum, 4, 8);
    float rinv = 1.0f / sum;
    for (int c = seg * 64; c < seg * 64 + 64; ++c)
      pr[row][c] = f2bf(lg[row][c] * rinv);
  }
  __syncthreads();

  // --- attn @ V: 2 d-tiles x 2 K-halves across 4 waves, 8 WMMAs each ---
  int t     = wave & 1;
  int khalf = wave >> 1;
  int dcol  = t * 16 + lrow;
  v8f c = zero8();
  for (int kt = 0; kt < 8; ++kt) {
    int mbase = khalf * 256 + kt * 32 + kb8;
    v16bf a = frag_from_runs(&pr[lrow][mbase], &pr[lrow][mbase + 16]);
    v16bf b = frag_from_runs(vh + (size_t)dcol * NCELLS + mbase,
                             vh + (size_t)dcol * NCELLS + mbase + 16);
    c = __builtin_amdgcn_wmma_f32_16x16x32_bf16(false, a, false, b,
                                                (short)0, c, false, false);
  }
#pragma unroll
  for (int r = 0; r < 8; ++r) part[wave][r + 8 * hi][lrow] = c[r];
  __syncthreads();

  // --- combine K-halves, write attended bf16 [bt][cell][h*32+d] ---
  int bt = bh >> 3;
  int h  = bh & 7;
  for (int j = 0; j < 4; ++j) {
    int o   = threadIdx.x * 4 + j;       // 0..511 over 16x32 outputs
    int row = o >> 5;
    int cc  = o & 31;
    int tt  = cc >> 4;
    int cl  = cc & 15;
    float v = part[tt][row][cl] + part[tt + 2][row][cl];
    int cell = n0 + row;
    att[((size_t)(bt * NCELLS + cell)) * DMODEL + h * HDIM + cc] = f2bf(v);
  }
}

// ---------------------------------------------------------------------------
// Kernel 4: output projection, f32 result.
// ---------------------------------------------------------------------------
__global__ void out_proj(const bf16* __restrict__ att, const bf16* __restrict__ WoT,
                         const float* __restrict__ bo, float* __restrict__ out) {
  int wave = threadIdx.x >> 5;
  int lane = threadIdx.x & 31;
  int tile = blockIdx.x * 4 + wave;      // 0 .. 65535
  int rowtile = tile >> 4;
  int coltile = tile & 15;
  int row0 = rowtile * 16;
  int lrow = lane & 15;
  int hi   = lane >> 4;
  int kb8  = hi * 8;
  int col  = coltile * 16 + lrow;

  const bf16* arow = att + (size_t)(row0 + lrow) * DMODEL;
  const bf16* wrow = WoT + (size_t)col * DMODEL;

  v8f acc = zero8();
#pragma unroll
  for (int kc = 0; kc < 8; ++kc) {
    int k0 = kc * 32 + kb8;
    v16bf a = frag_from_runs(arow + k0, arow + k0 + 16);
    v16bf b = frag_from_runs(wrow + k0, wrow + k0 + 16);
    acc = __builtin_amdgcn_wmma_f32_16x16x32_bf16(false, a, false, b,
                                                  (short)0, acc, false, false);
  }
  float bval = bo[col];
#pragma unroll
  for (int r = 0; r < 8; ++r) {
    int grow = row0 + r + 8 * hi;
    out[(size_t)grow * DMODEL + col] = acc[r] + bval;
  }
}

// ---------------------------------------------------------------------------
extern "C" void kernel_launch(void* const* d_in, const int* in_sizes, int n_in,
                              void* d_out, int out_size, void* d_ws, size_t ws_size,
                              hipStream_t stream) {
  (void)in_sizes; (void)n_in; (void)out_size; (void)ws_size;
  const float* x  = (const float*)d_in[0];
  const float* sc = (const float*)d_in[1];
  const float* Wq = (const float*)d_in[2];
  const float* bq = (const float*)d_in[3];
  const float* Wk = (const float*)d_in[4];
  const float* bk = (const float*)d_in[5];
  const float* Wv = (const float*)d_in[6];
  const float* bv = (const float*)d_in[7];
  const float* Wo = (const float*)d_in[8];
  const float* bo = (const float*)d_in[9];
  float* out = (float*)d_out;

  char* ws = (char*)d_ws;
  bf16* WT  = (bf16*)ws;                               // [4][256][256] bf16 (q,k,v,o)
  size_t off = (size_t)4 * DMODEL * DMODEL * 2;        // 512 KB
  const size_t XB_BYTES  = (size_t)BT * NCELLS * DMODEL * 2;          // 32 MB
  const size_t QKV_BYTES = (size_t)BT * NHEADS * NCELLS * HDIM * 2;   // 32 MB
  bf16* xbuf = (bf16*)(ws + off); off += XB_BYTES;
  bf16* qb   = (bf16*)(ws + off); off += QKV_BYTES;
  bf16* kb   = (bf16*)(ws + off); off += QKV_BYTES;
  bf16* vtb  = (bf16*)(ws + off); off += QKV_BYTES;
  bf16* att  = (bf16*)(ws + off);                      // [BT*NCELLS, DMODEL] bf16

  // x has BT*NCELLS*DMODEL = 16,777,216 elems; 8 per thread, 256 per block.
  conv_x      <<<8192, 256, 0, stream>>>(x, xbuf);
  prep_weights<<<1024, 256, 0, stream>>>(Wq, Wk, Wv, Wo, WT);
  qkv_proj    <<<49152, 128, 0, stream>>>(xbuf, WT, bq, bk, bv, qb, kb, vtb);
  attn_kernel <<<32768, 128, 0, stream>>>(qb, kb, vtb, sc, att);
  out_proj    <<<16384, 128, 0, stream>>>(att, WT + 3 * DMODEL * DMODEL, bo, out);
}
